// WindowAttention_57458072486515
// MI455X (gfx1250) — matrix operand order repaired and checked by reading, not compiled, not run on hardware
//
#include <hip/hip_runtime.h>

typedef __attribute__((ext_vector_type(16))) _Float16 v16h;
typedef __attribute__((ext_vector_type(8)))  _Float16 v8h;
typedef __attribute__((ext_vector_type(8)))  float    v8f;

#define NT    49     // tokens per window
#define NPAD  64     // padded tokens
#define CDIM  128    // channels
#define HD    32     // head dim
#define NH    4      // heads
#define WAVES 8      // 256 threads / wave32

// ---- WMMA fragment loaders (wave32 CDNA5 layouts, CDNA5 ISA) ----
// A (16x32 f16): lane M=l&15, g=l>>4 ; halfs 0..15 -> K=g*16..g*16+15 (contiguous, matches B)
__device__ __forceinline__ v16h load_a16(const _Float16* A, int lda, int m0, int k0, int lane) {
  const int M = lane & 15, g = lane >> 4;
  const _Float16* p = A + (size_t)(m0 + M) * lda + k0 + g * 16;
  v8h lo = *(const v8h*)p;
  v8h hi = *(const v8h*)(p + 8);
  v16h r;
#pragma unroll
  for (int i = 0; i < 8; ++i) { r[i] = lo[i]; r[i + 8] = hi[i]; }
  return r;
}

// B (32x16 f16), from K-major storage Bt[n][k]: lane n=l&15 fixed, 16 contiguous K at (l>>4)*16
__device__ __forceinline__ v16h load_b16(const _Float16* Bt, int ldb, int n0, int k0, int lane) {
  const _Float16* p = Bt + (size_t)(n0 + (lane & 15)) * ldb + k0 + (lane >> 4) * 16;
  return *(const v16h*)p;
}

// 16x16 output tile, K=128 (4 chained WMMAs)
__device__ __forceinline__ v8f gemm_k128(const _Float16* A, int lda, int m0,
                                         const _Float16* Bt, int ldb, int n0, int lane) {
  v8f acc = {};
#pragma unroll
  for (int kk = 0; kk < 4; ++kk) {
    v16h a  = load_a16(A, lda, m0, kk * 32, lane);
    v16h bm = load_b16(Bt, ldb, n0, kk * 32, lane);
    acc = __builtin_amdgcn_wmma_f32_16x16x32_f16(false, a, false, bm, (short)0, acc, false, false);
  }
  return acc;
}

// ---- prologue: fp32 weights -> K-major f16 copies in workspace ----
__global__ void convert_weights(const float* __restrict__ wqkv, const float* __restrict__ wproj,
                                _Float16* __restrict__ wqkvT, _Float16* __restrict__ wprojT) {
  int i = blockIdx.x * blockDim.x + threadIdx.x;
  if (i < 384 * 128) {                 // wqkvT[n][k] = wqkv[k][n], wqkv is (128,384)
    int n = i >> 7, k = i & 127;
    wqkvT[i] = (_Float16)wqkv[(size_t)k * 384 + n];
  }
  if (i < 128 * 128) {                 // wprojT[n][k] = wproj[k][n], wproj is (128,128)
    int n = i >> 7, k = i & 127;
    wprojT[i] = (_Float16)wproj[(size_t)k * 128 + n];
  }
}

// ---- fused window attention: one block per window ----
__global__ __launch_bounds__(256)
void win_attn_kernel(const float* __restrict__ x, const float* __restrict__ mask,
                     const _Float16* __restrict__ wqkvT, const float* __restrict__ bqkv,
                     const _Float16* __restrict__ wprojT, const float* __restrict__ bproj,
                     float* __restrict__ out) {
  extern __shared__ __align__(32) char smem[];
  _Float16* xs = (_Float16*)(smem);            // [64][128] staged x (f16); reused for mask after QKV
  _Float16* qs = (_Float16*)(smem + 16384);    // [64][128] Q
  _Float16* ks = (_Float16*)(smem + 32768);    // [64][128] K
  _Float16* vt = (_Float16*)(smem + 49152);    // [128][64] V transposed (dim-major)
  _Float16* ao = (_Float16*)(smem + 65536);    // [64][128] attention output (pre-proj)
  _Float16* ps = (_Float16*)(smem + 81920);    // [64][64]  softmax probs (f16)
  float*    ss = (float*)   (smem + 90112);    // [64][68]  attention logits (f32)

  const int b    = blockIdx.x;
  const int tid  = threadIdx.x;
  const int wave = tid >> 5, lane = tid & 31;
  const int ln   = lane & 15, lg = lane >> 4;

  // pull the L2-resident f16 weight images toward this WGP
  __builtin_prefetch(wqkvT + (size_t)tid * 192, 0, 1);
  __builtin_prefetch(wprojT + (size_t)tid * 64, 0, 1);

  // ---- stage x -> LDS f16, zero-pad rows >= 49 ----
  const float* xb = x + (size_t)b * NT * CDIM;
  for (int i = tid; i < NPAD * CDIM; i += 256) {
    int r = i >> 7;
    xs[i] = (r < NT) ? (_Float16)xb[(size_t)r * CDIM + (i & 127)] : (_Float16)0.f;
  }
  __syncthreads();

  // ---- QKV = x @ w_qkv + b : three uniform passes (no divergent epilogue) ----
  // Q tiles: 4 row-tiles x 8 col-tiles
  for (int t = wave; t < 32; t += WAVES) {
    int m0 = (t & 3) * 16, n0 = (t >> 2) * 16;
    v8f acc = gemm_k128(xs, CDIM, m0, wqkvT, CDIM, n0, lane);
    int n = n0 + ln;
    float bias = bqkv[n];                       // SCALE == 1.0
#pragma unroll
    for (int r = 0; r < 8; ++r)
      qs[(m0 + r + lg * 8) * CDIM + n] = (_Float16)(acc[r] + bias);
  }
  // K tiles
  for (int t = wave; t < 32; t += WAVES) {
    int m0 = (t & 3) * 16, n0 = (t >> 2) * 16;
    v8f acc = gemm_k128(xs, CDIM, m0, wqkvT + 128 * CDIM, CDIM, n0, lane);
    int n = n0 + ln;
    float bias = bqkv[128 + n];
#pragma unroll
    for (int r = 0; r < 8; ++r)
      ks[(m0 + r + lg * 8) * CDIM + n] = (_Float16)(acc[r] + bias);
  }
  // V tiles -> stored transposed; token index contiguous per lane -> one b128 store
  for (int t = wave; t < 32; t += WAVES) {
    int m0 = (t & 3) * 16, n0 = (t >> 2) * 16;
    v8f acc = gemm_k128(xs, CDIM, m0, wqkvT + 256 * CDIM, CDIM, n0, lane);
    int n = n0 + ln;
    float bias = bqkv[256 + n];
    v8h pk;
#pragma unroll
    for (int r = 0; r < 8; ++r) pk[r] = (_Float16)(acc[r] + bias);
    *(v8h*)(vt + (size_t)n * NPAD + m0 + lg * 8) = pk;
  }
  __syncthreads();

  // ---- stage this window's mask into LDS (xs is dead now) ----
  float* mlds = (float*)xs;                       // 49*49*4 = 9604 B <= 16384 B
  const float* mk = mask + (size_t)(b & 63) * NT * NT;   // window id = b % nW
  for (int i = tid; i < NT * NT; i += 256) mlds[i] = mk[i];
  __syncthreads();

  for (int h = 0; h < NH; ++h) {
    // ---- S = Q_h @ K_h^T : 4x4 tiles, K=32 (one WMMA each) ----
    for (int t = wave; t < 16; t += WAVES) {
      int m0 = (t >> 2) * 16, n0 = (t & 3) * 16;
      v16h a  = load_a16(qs + h * HD, CDIM, m0, 0, lane);
      v16h bm = load_b16(ks + h * HD, CDIM, n0, 0, lane);
      v8f acc = {};
      acc = __builtin_amdgcn_wmma_f32_16x16x32_f16(false, a, false, bm, (short)0, acc, false, false);
      int n = n0 + ln;
#pragma unroll
      for (int r = 0; r < 8; ++r) {
        int m = m0 + r + lg * 8;
        float v = (m < NT && n < NT) ? (acc[r] + mlds[m * NT + n]) : -1e30f;
        ss[m * 68 + n] = v;
      }
    }
    __syncthreads();

    // ---- softmax: one row per thread (rows 0..63) ----
    if (tid < NPAD) {
      const float* row = ss + tid * 68;
      float mx = -1e30f;
      for (int c = 0; c < NT; ++c) mx = fmaxf(mx, row[c]);
      float sum = 0.f;
      for (int c = 0; c < NT; ++c) sum += __expf(row[c] - mx);
      float inv = 1.f / sum;
      _Float16* prow = ps + tid * 64;
      for (int c = 0; c < NT; ++c) prow[c] = (_Float16)(__expf(row[c] - mx) * inv);
      for (int c = NT; c < 64; ++c) prow[c] = (_Float16)0.f;
    }
    __syncthreads();

    // ---- out_h = P @ V_h : 4x2 tiles, K=64 (2 WMMA); exactly one tile per wave ----
    {
      int m0 = (wave >> 1) * 16, n0 = (wave & 1) * 16;
      v8f acc = {};
#pragma unroll
      for (int kk = 0; kk < 2; ++kk) {
        v16h a  = load_a16(ps, 64, m0, kk * 32, lane);
        v16h bm = load_b16(vt + (size_t)h * HD * NPAD, NPAD, n0, kk * 32, lane);
        acc = __builtin_amdgcn_wmma_f32_16x16x32_f16(false, a, false, bm, (short)0, acc, false, false);
      }
      int n = h * HD + n0 + ln;
#pragma unroll
      for (int r = 0; r < 8; ++r)
        ao[(m0 + r + lg * 8) * CDIM + n] = (_Float16)acc[r];
    }
    __syncthreads();   // protect ss/ps reuse in next head
  }

  // ---- final: out = ao @ w_proj + b_proj : 4x8 tiles, K=128 (4 WMMA each) ----
  float* ob = out + (size_t)b * NT * CDIM;
  for (int t = wave; t < 32; t += WAVES) {
    int m0 = (t & 3) * 16, n0 = (t >> 2) * 16;
    v8f acc = gemm_k128(ao, CDIM, m0, wprojT, CDIM, n0, lane);
    int n = n0 + ln;
    float bias = bproj[n];
#pragma unroll
    for (int r = 0; r < 8; ++r) {
      int m = m0 + r + lg * 8;
      if (m < NT) ob[(size_t)m * CDIM + n] = acc[r] + bias;
    }
  }
}

extern "C" void kernel_launch(void* const* d_in, const int* in_sizes, int n_in,
                              void* d_out, int out_size, void* d_ws, size_t ws_size,
                              hipStream_t stream) {
  const float* x     = (const float*)d_in[0];
  const float* mask  = (const float*)d_in[1];
  const float* wqkv  = (const float*)d_in[2];
  const float* bqkv  = (const float*)d_in[3];
  const float* wproj = (const float*)d_in[4];
  const float* bproj = (const float*)d_in[5];
  float* out = (float*)d_out;

  _Float16* wqkvT  = (_Float16*)d_ws;                              // 384*128*2 = 98304 B
  _Float16* wprojT = (_Float16*)((char*)d_ws + 384 * 128 * 2);     // 128*128*2 = 32768 B

  convert_weights<<<(384 * 128 + 255) / 256, 256, 0, stream>>>(wqkv, wproj, wqkvT, wprojT);

  const size_t shmem = 107520;   // xs/qs/ks/vt/ao/ps/ss arena (fits 320KB/WGP)
  win_attn_kernel<<<4096, 256, shmem, stream>>>(x, mask, wqkvT, bqkv, wprojT, bproj, out);
}
